// MLPDeepSeekV3_5282809775019
// MI455X (gfx1250) — 8-bit matrix operand order repaired and checked by reading, compiled, not run on hardware
//
#include <hip/hip_runtime.h>
#include <hip/hip_bf16.h>

typedef __attribute__((ext_vector_type(16))) int   v16i;
typedef __attribute__((ext_vector_type(8)))  float v8f;

union AFrag { v16i v; unsigned long long u[8]; };
union BFrag { v16i v; int4 q[4]; };

__device__ __forceinline__ unsigned char f32_to_fp8_sw(float f) {
    unsigned int u = __float_as_uint(f);
    unsigned char s = (unsigned char)((u >> 24) & 0x80u);
    float a = fabsf(f);
    if (!(a < 448.f)) return (unsigned char)(s | 0x7Eu);
    unsigned int au  = u & 0x7FFFFFFFu;
    int exp          = (int)(au >> 23) - 127;
    unsigned int man = au & 0x7FFFFFu;
    if (exp < -10) return s;
    if (exp >= -6) {
        unsigned int r   = man >> 20;
        unsigned int rem = man & 0xFFFFFu;
        r += (rem > 0x80000u) || (rem == 0x80000u && (r & 1u));
        unsigned int val = ((unsigned int)(exp + 7) << 3) + r;
        if (val > 0x7Eu) val = 0x7Eu;
        return (unsigned char)(s | val);
    }
    unsigned int full = 0x800000u | man;
    int shr           = 14 - exp;
    unsigned int r    = full >> shr;
    unsigned int rem  = full & ((1u << shr) - 1u);
    unsigned int half = 1u << (shr - 1);
    r += (rem > half) || (rem == half && (r & 1u));
    return (unsigned char)(s | r);
}

__device__ __forceinline__ unsigned int pack4(float a, float b, float c, float d, float inv) {
#if __has_builtin(__builtin_amdgcn_cvt_pk_fp8_f32)
    float x0 = fminf(fmaxf(a * inv, -448.f), 448.f);
    float x1 = fminf(fmaxf(b * inv, -448.f), 448.f);
    float x2 = fminf(fmaxf(c * inv, -448.f), 448.f);
    float x3 = fminf(fmaxf(d * inv, -448.f), 448.f);
    int lo = __builtin_amdgcn_cvt_pk_fp8_f32(x0, x1, 0,  false);
    int r  = __builtin_amdgcn_cvt_pk_fp8_f32(x2, x3, lo, true);
    return (unsigned int)r;
#else
    return (unsigned int)f32_to_fp8_sw(a * inv)
         | ((unsigned int)f32_to_fp8_sw(b * inv) << 8)
         | ((unsigned int)f32_to_fp8_sw(c * inv) << 16)
         | ((unsigned int)f32_to_fp8_sw(d * inv) << 24);
#endif
}

__global__ __launch_bounds__(256) void k_quant_w(const float* __restrict__ w,
                                                 unsigned char* __restrict__ q) {
    size_t gid = (size_t)blockIdx.x * 256 + threadIdx.x;
    const float4* p = (const float4*)w + gid * 2;
    float4 v0 = p[0], v1 = p[1];
    unsigned int lo = pack4(v0.x, v0.y, v0.z, v0.w, 1.f);
    unsigned int hi = pack4(v1.x, v1.y, v1.z, v1.w, 1.f);
    ((unsigned long long*)q)[gid] = (unsigned long long)lo | ((unsigned long long)hi << 32);
}

__global__ __launch_bounds__(256) void k_quant_act(const float* __restrict__ x,
                                                   unsigned char* __restrict__ q,
                                                   float* __restrict__ s) {
    int wid  = blockIdx.x * 8 + (threadIdx.x >> 5);
    int lane = threadIdx.x & 31;
    float4 v = ((const float4*)x)[(size_t)wid * 32 + lane];
    float m = fmaxf(fmaxf(fabsf(v.x), fabsf(v.y)), fmaxf(fabsf(v.z), fabsf(v.w)));
#pragma unroll
    for (int d = 16; d; d >>= 1) m = fmaxf(m, __shfl_xor(m, d));
    float scale = (m == 0.f) ? 1.f : m * (1.f / 448.f);
    float inv   = 1.f / scale;
    ((unsigned int*)q)[(size_t)wid * 32 + lane] = pack4(v.x, v.y, v.z, v.w, inv);
    if (lane == 0) s[wid] = scale;
}

__global__ __launch_bounds__(256) void k_gemm1_swiglu(
    const unsigned char* __restrict__ A8, const float* __restrict__ As,
    const unsigned char* __restrict__ W1, const float* __restrict__ S1,
    const unsigned char* __restrict__ W3, const float* __restrict__ S3,
    unsigned char* __restrict__ H8, float* __restrict__ Hs,
    int K, int KB, int NOUT) {
    extern __shared__ char smem[];
    unsigned char* smA  = (unsigned char*)smem;
    unsigned char* smB1 = smA + 16384;
    unsigned char* smB3 = smB1 + 16384;
    float*         sSc  = (float*)(smB3 + 16384);

    const int tid  = threadIdx.x;
    const int t0   = blockIdx.x * 128;
    const int n0   = blockIdx.y * 128;
    const int w    = tid >> 5, lane = tid & 31;
    const int wy   = w >> 1,   wx   = w & 1;
    const int hi   = lane >> 4, ln  = lane & 15;
    const int row  = tid >> 1, half = tid & 1;

    const unsigned char* gA = A8 + (size_t)(t0 + row) * K + half * 64;
    const unsigned char* g1 = W1 + (size_t)(n0 + row) * K + half * 64;
    const unsigned char* g3 = W3 + (size_t)(n0 + row) * K + half * 64;
    int4* lA = (int4*)(smA  + row * 128 + half * 64);
    int4* l1 = (int4*)(smB1 + row * 128 + half * 64);
    int4* l3 = (int4*)(smB3 + row * 128 + half * 64);

    v8f acc1[2][4], acc3[2][4];
#pragma unroll
    for (int mi = 0; mi < 2; ++mi)
#pragma unroll
        for (int ni = 0; ni < 4; ++ni)
#pragma unroll
            for (int i = 0; i < 8; ++i) { acc1[mi][ni][i] = 0.f; acc3[mi][ni][i] = 0.f; }

    int4 ra[4], rb1[4], rb3[4];
    float rsc = 0.f;
#pragma unroll
    for (int j = 0; j < 4; ++j) {
        ra[j]  = ((const int4*)gA)[j];
        rb1[j] = ((const int4*)g1)[j];
        rb3[j] = ((const int4*)g3)[j];
    }
    if (tid < 128) rsc = As[(size_t)(t0 + tid) * KB + 0];

    for (int kb = 0; kb < KB; ++kb) {
#pragma unroll
        for (int j = 0; j < 4; ++j) { lA[j] = ra[j]; l1[j] = rb1[j]; l3[j] = rb3[j]; }
        if (tid < 128) sSc[tid] = rsc;
        __syncthreads();

        if (kb + 1 < KB) {
            const size_t koff = (size_t)(kb + 1) * 128;
#pragma unroll
            for (int j = 0; j < 4; ++j) {
                ra[j]  = ((const int4*)(gA + koff))[j];
                rb1[j] = ((const int4*)(g1 + koff))[j];
                rb3[j] = ((const int4*)(g3 + koff))[j];
            }
            if (tid < 128) rsc = As[(size_t)(t0 + tid) * KB + kb + 1];
        }

        const float sB1 = S1[(size_t)blockIdx.y * KB + kb];
        const float sB3 = S3[(size_t)blockIdx.y * KB + kb];

        AFrag a[2];
#pragma unroll
        for (int mi = 0; mi < 2; ++mi) {
            const unsigned long long* ap =
                (const unsigned long long*)(smA + (wy * 32 + mi * 16 + ln) * 128);
#pragma unroll
            for (int c = 0; c < 8; ++c) a[mi].u[c] = ap[(c >> 1) * 4 + hi * 2 + (c & 1)];
        }
        v8f f1[2], f3[2];
#pragma unroll
        for (int mi = 0; mi < 2; ++mi) {
            const float* sp = &sSc[wy * 32 + mi * 16 + hi * 8];
#pragma unroll
            for (int i = 0; i < 8; ++i) { f1[mi][i] = sp[i] * sB1; f3[mi][i] = sp[i] * sB3; }
        }
        const v8f z = {0.f, 0.f, 0.f, 0.f, 0.f, 0.f, 0.f, 0.f};
#pragma unroll
        for (int ni = 0; ni < 4; ++ni) {
            const int nl = wx * 64 + ni * 16 + ln;
            BFrag b1, b3;
#pragma unroll
            for (int j = 0; j < 4; ++j) {
                b1.q[j] = *(const int4*)(smB1 + nl * 128 + j * 32 + hi * 16);
                b3.q[j] = *(const int4*)(smB3 + nl * 128 + j * 32 + hi * 16);
            }
#pragma unroll
            for (int mi = 0; mi < 2; ++mi) {
                v8f p1 = __builtin_amdgcn_wmma_f32_16x16x128_fp8_fp8(
                    a[mi].v, b1.v, (short)0, z, false, false);
                acc1[mi][ni] += p1 * f1[mi];
                v8f p3 = __builtin_amdgcn_wmma_f32_16x16x128_fp8_fp8(
                    a[mi].v, b3.v, (short)0, z, false, false);
                acc3[mi][ni] += p3 * f3[mi];
            }
        }
        __syncthreads();
    }

    float* ht = (float*)smem;
    const int P = 132;
#pragma unroll
    for (int mi = 0; mi < 2; ++mi)
#pragma unroll
        for (int ni = 0; ni < 4; ++ni)
#pragma unroll
            for (int i = 0; i < 8; ++i) {
                float g  = acc1[mi][ni][i];
                float hv = (g / (1.f + __expf(-g))) * acc3[mi][ni][i];
                int m = wy * 32 + mi * 16 + i + 8 * hi;
                int n = wx * 64 + ni * 16 + ln;
                ht[m * P + n] = hv;
            }
    __syncthreads();
    {
        const int r = tid >> 1, hf = tid & 1;
        const float4* rp = (const float4*)(&ht[r * P + hf * 64]);
        float4 vv[16];
        float m = 0.f;
#pragma unroll
        for (int j = 0; j < 16; ++j) {
            vv[j] = rp[j];
            m = fmaxf(m, fmaxf(fmaxf(fabsf(vv[j].x), fabsf(vv[j].y)),
                               fmaxf(fabsf(vv[j].z), fabsf(vv[j].w))));
        }
        m = fmaxf(m, __shfl_xor(m, 1));
        float scale = (m == 0.f) ? 1.f : m * (1.f / 448.f);
        float inv   = 1.f / scale;
        unsigned long long* gq =
            (unsigned long long*)(H8 + (size_t)(t0 + r) * NOUT + n0 + hf * 64);
#pragma unroll
        for (int j = 0; j < 8; ++j) {
            unsigned int lo  = pack4(vv[2*j].x,   vv[2*j].y,   vv[2*j].z,   vv[2*j].w,   inv);
            unsigned int hiw = pack4(vv[2*j+1].x, vv[2*j+1].y, vv[2*j+1].z, vv[2*j+1].w, inv);
            gq[j] = (unsigned long long)lo | ((unsigned long long)hiw << 32);
        }
        if (hf == 0) Hs[(size_t)(t0 + r) * gridDim.y + blockIdx.y] = scale;
    }
}

__global__ __launch_bounds__(256) void k_gemm_out(
    const unsigned char* __restrict__ A8, const float* __restrict__ As,
    const unsigned char* __restrict__ Wq, const float* __restrict__ Ws,
    float* __restrict__ C, int K, int KB, int N) {
    extern __shared__ char smem[];
    unsigned char* smA = (unsigned char*)smem;
    unsigned char* smB = smA + 16384;
    float*         sSc = (float*)(smB + 16384);

    const int tid  = threadIdx.x;
    const int t0   = blockIdx.x * 128;
    const int n0   = blockIdx.y * 128;
    const int w    = tid >> 5, lane = tid & 31;
    const int wy   = w >> 1,   wx   = w & 1;
    const int hi   = lane >> 4, ln  = lane & 15;
    const int row  = tid >> 1, half = tid & 1;

    const unsigned char* gA = A8 + (size_t)(t0 + row) * K + half * 64;
    const unsigned char* gB = Wq + (size_t)(n0 + row) * K + half * 64;
    int4* lA = (int4*)(smA + row * 128 + half * 64);
    int4* lB = (int4*)(smB + row * 128 + half * 64);

    v8f acc[2][4];
#pragma unroll
    for (int mi = 0; mi < 2; ++mi)
#pragma unroll
        for (int ni = 0; ni < 4; ++ni)
#pragma unroll
            for (int i = 0; i < 8; ++i) acc[mi][ni][i] = 0.f;

    int4 ra[4], rb[4];
    float rsc = 0.f;
#pragma unroll
    for (int j = 0; j < 4; ++j) { ra[j] = ((const int4*)gA)[j]; rb[j] = ((const int4*)gB)[j]; }
    if (tid < 128) rsc = As[(size_t)(t0 + tid) * KB + 0];

    for (int kb = 0; kb < KB; ++kb) {
#pragma unroll
        for (int j = 0; j < 4; ++j) { lA[j] = ra[j]; lB[j] = rb[j]; }
        if (tid < 128) sSc[tid] = rsc;
        __syncthreads();

        if (kb + 1 < KB) {
            const size_t koff = (size_t)(kb + 1) * 128;
#pragma unroll
            for (int j = 0; j < 4; ++j) {
                ra[j] = ((const int4*)(gA + koff))[j];
                rb[j] = ((const int4*)(gB + koff))[j];
            }
            if (tid < 128) rsc = As[(size_t)(t0 + tid) * KB + kb + 1];
        }

        const float sB = Ws[(size_t)blockIdx.y * KB + kb];

        AFrag a[2];
#pragma unroll
        for (int mi = 0; mi < 2; ++mi) {
            const unsigned long long* ap =
                (const unsigned long long*)(smA + (wy * 32 + mi * 16 + ln) * 128);
#pragma unroll
            for (int c = 0; c < 8; ++c) a[mi].u[c] = ap[(c >> 1) * 4 + hi * 2 + (c & 1)];
        }
        v8f f[2];
#pragma unroll
        for (int mi = 0; mi < 2; ++mi) {
            const float* sp = &sSc[wy * 32 + mi * 16 + hi * 8];
#pragma unroll
            for (int i = 0; i < 8; ++i) f[mi][i] = sp[i] * sB;
        }
        const v8f z = {0.f, 0.f, 0.f, 0.f, 0.f, 0.f, 0.f, 0.f};
#pragma unroll
        for (int ni = 0; ni < 4; ++ni) {
            const int nl = wx * 64 + ni * 16 + ln;
            BFrag b;
#pragma unroll
            for (int j = 0; j < 4; ++j)
                b.q[j] = *(const int4*)(smB + nl * 128 + j * 32 + hi * 16);
#pragma unroll
            for (int mi = 0; mi < 2; ++mi) {
                v8f p = __builtin_amdgcn_wmma_f32_16x16x128_fp8_fp8(
                    a[mi].v, b.v, (short)0, z, false, false);
                acc[mi][ni] += p * f[mi];
            }
        }
        __syncthreads();
    }

#pragma unroll
    for (int mi = 0; mi < 2; ++mi)
#pragma unroll
        for (int ni = 0; ni < 4; ++ni)
#pragma unroll
            for (int i = 0; i < 8; ++i) {
                int m = t0 + wy * 32 + mi * 16 + i + 8 * hi;
                int n = n0 + wx * 64 + ni * 16 + ln;
                C[(size_t)m * N + n] = acc[mi][ni][i];
            }
}

extern "C" void kernel_launch(void* const* d_in, const int* in_sizes, int n_in,
                              void* d_out, int out_size, void* d_ws, size_t ws_size,
                              hipStream_t stream) {
    constexpr int T = 2048, DIM = 4096, INTER = 11008;
    constexpr int KB1 = DIM / 128;
    constexpr int KB2 = INTER / 128;

    const float* x   = (const float*)d_in[0];
    const float* w1q = (const float*)d_in[1];
    const float* w1s = (const float*)d_in[2];
    const float* w2q = (const float*)d_in[3];
    const float* w2s = (const float*)d_in[4];
    const float* w3q = (const float*)d_in[5];
    const float* w3s = (const float*)d_in[6];
    float* out = (float*)d_out;

    char* ws = (char*)d_ws;
    size_t off = 0;
    auto alloc = [&](size_t bytes) -> char* {
        char* p = ws + off;
        off = (off + bytes + 255) & ~(size_t)255;
        return p;
    };
    unsigned char* x8   = (unsigned char*)alloc((size_t)T * DIM);
    float*         xs   = (float*)alloc((size_t)T * KB1 * 4);
    unsigned char* h8   = (unsigned char*)alloc((size_t)T * INTER);
    float*         hs   = (float*)alloc((size_t)T * KB2 * 4);
    unsigned char* w1_8 = (unsigned char*)alloc((size_t)INTER * DIM);
    unsigned char* w3_8 = (unsigned char*)alloc((size_t)INTER * DIM);
    unsigned char* w2_8 = (unsigned char*)alloc((size_t)DIM * INTER);
    (void)ws_size; (void)in_sizes; (void)n_in; (void)out_size;

    const int wgW = (int)(((size_t)INTER * DIM / 8) / 256);
    k_quant_w<<<wgW, 256, 0, stream>>>(w1q, w1_8);
    k_quant_w<<<wgW, 256, 0, stream>>>(w3q, w3_8);
    k_quant_w<<<wgW, 256, 0, stream>>>(w2q, w2_8);

    k_quant_act<<<(T * KB1) / 8, 256, 0, stream>>>(x, x8, xs);

    k_gemm1_swiglu<<<dim3(T / 128, INTER / 128), 256, 132 * 128 * 4, stream>>>(
        x8, xs, w1_8, w1s, w3_8, w3s, h8, hs, DIM, KB1, INTER);

    k_gemm_out<<<dim3(T / 128, DIM / 128), 256, 16384 * 2 + 512, stream>>>(
        h8, hs, w2_8, w2s, out, INTER, KB2, DIM);
}
